// ReluNetworkDisentangle_76510547411465
// MI455X (gfx1250) — compile-verified, matrix operand order checked by reading, not run on hardware
//
#include <hip/hip_runtime.h>
#include <math.h>

typedef __attribute__((ext_vector_type(16))) _Float16 v16h;
typedef __attribute__((ext_vector_type(4)))  _Float16 v4h;
typedef __attribute__((ext_vector_type(8)))  float    v8f;

#define HH     128   // hidden width
#define NBLK   8     // residual blocks
#define WAVES  4     // waves per workgroup (wave32)
#define PTSW   16    // points per wave tile

// f16 B-fragment workspace layout (offsets in halfs). A K x N matrix is stored
// as frag[t][c][lane(32)][e(16)]: one 32-byte contiguous v16h per lane.
#define OFF_WIN     0        // 64x128, row63 = b_in+b_c        ->  8192 halfs
#define OFF_WC      8192     // 128x128                         -> 16384
#define OFF_WSKIP   24576    // 64x128, row63 = b_skip+b_skipc  ->  8192
#define OFF_WSKIPC  32768    // 128x128                         -> 16384
#define OFF_WVIEW   49152    // 32x128, row27 = b_view+b_viewc  ->  4096
#define OFF_WVIEWC  53248    // 128x128                         -> 16384
#define OFF_WBLK    69632    // 8 x 128x128                     -> 131072
#define OFF_WOUT    200704   // 128x16 head (col0=W_out)        ->  2048
#define OFF_WRGB    202752   // 128x16 head (col0..2=W_rgb)     ->  2048
#define WS_HALFS    204800   // 409600 bytes

// K index striping for 16-bit A/B WMMA fragments (wave32), per CDNA5 ISA 7.12.2
__device__ __forceinline__ int kmap(int e, int h) {
  return (e < 8) ? (h * 8 + e) : (16 + h * 8 + (e - 8));
}

// ---------------- pre-pass: f32 weights -> f16 WMMA B-fragments ----------------
struct SwzJob {
  const float* W; const float* bias0; const float* bias1;
  int dstOff; int krows; int nchunks; int ntiles; int ldn; int ncols;
};
struct SwzParams { SwzJob jobs[16]; _Float16* ws; };

__global__ __launch_bounds__(256) void swizzle_weights_kernel(SwzParams S) {
  SwzJob job = S.jobs[blockIdx.y];
  int idx = blockIdx.x * blockDim.x + threadIdx.x;
  int total = job.ntiles * job.nchunks * 512;   // t * c * lane(32) * e(16)
  if (idx >= total) return;
  int e    = idx & 15;
  int lane = (idx >> 4) & 31;
  int rest = idx >> 9;                          // t*nchunks + c
  int c    = rest % job.nchunks;
  int t    = rest / job.nchunks;
  int h    = lane >> 4;
  int n    = t * 16 + (lane & 15);
  int k    = c * 32 + kmap(e, h);
  float v  = 0.0f;
  if (n < job.ncols) {
    if (k < job.krows)                 v = job.W[k * job.ldn + n];
    else if (k == job.krows && job.bias0)
      v = job.bias0[n] + (job.bias1 ? job.bias1[n] : 0.0f);
  }
  S.ws[job.dstOff + idx] = (_Float16)v;
}

// ---------------- fused MLP ----------------
struct NerfParams {
  const float* coords;     // (N,3)
  const float* condition;  // (N,256)
  const float* ray_dir;    // (N,3)
  const float* noise;      // (N,1)
  const float* b_blocks;   // (8,128)
  const float* b_out;      // (1)
  const float* b_rgb;      // (3)
  const _Float16* ws;      // pre-swizzled f16 B-fragments
  float* out;              // (N,4)
};

// A fragment (16xK tile, row-major f16 in LDS): row m = lane&15
__device__ __forceinline__ v16h load_A_lds(const _Float16* tile, int ld, int kbase,
                                           int m, int h) {
  v16h a;
#pragma unroll
  for (int e = 0; e < 16; ++e) a[e] = tile[m * ld + kbase + kmap(e, h)];
  return a;
}

// B fragment: one aligned 32-byte vector load from the pre-swizzled workspace
__device__ __forceinline__ v16h load_B_ws(const _Float16* frags, int t, int c,
                                          int nchunks, int lane) {
  return *(const v16h*)(frags + (((size_t)(t * nchunks + c) * 32 + lane) << 4));
}

#define WMMA_F16(a, b, c) \
  __builtin_amdgcn_wmma_f32_16x16x32_f16(false, (a), false, (b), (short)0, (c), false, false)

__global__ __launch_bounds__(WAVES * 32, 1)
void nerf_fused_kernel(NerfParams P) {
  __shared__ _Float16 sEncP[WAVES][PTSW][64];   // 63 real + bias-row(=1.0)
  __shared__ _Float16 sEncV[WAVES][PTSW][32];   // 27 real + bias-row(=1.0) + zeros
  __shared__ _Float16 sCondS[WAVES][PTSW][HH];  // condition[:, :128]
  __shared__ _Float16 sCondA[WAVES][PTSW][HH];  // condition[:, 128:]
  __shared__ _Float16 sAct[WAVES][PTSW][HH];    // relu(net) bounce tile

  const int lane = threadIdx.x & 31;
  const int w    = threadIdx.x >> 5;
  const int ln   = lane & 15;
  const int h    = lane >> 4;
  const int p0   = (blockIdx.x * WAVES + w) * PTSW;

  // ---- stage condition halves: float4 loads, packed f16 stores ----
  {
#pragma unroll
    for (int i = 0; i < 16; ++i) {          // 512 float4 per matrix per wave
      int idx = i * 32 + lane;
      int r = idx >> 5, c4 = idx & 31;
      const float* row = P.condition + (size_t)(p0 + r) * 256 + c4 * 4;
      float4 vs = *(const float4*)row;
      float4 va = *(const float4*)(row + HH);
      *(v4h*)&sCondS[w][r][c4 * 4] =
          (v4h){(_Float16)vs.x, (_Float16)vs.y, (_Float16)vs.z, (_Float16)vs.w};
      *(v4h*)&sCondA[w][r][c4 * 4] =
          (v4h){(_Float16)va.x, (_Float16)va.y, (_Float16)va.z, (_Float16)va.w};
    }
  }

  // ---- positional encoding (L=10): slots 0..62, slot 63 = 1.0 (bias row) ----
  {
    int m = ln, p = p0 + m;
    float c3[3] = {P.coords[p * 3 + 0], P.coords[p * 3 + 1], P.coords[p * 3 + 2]};
#pragma unroll
    for (int jj = 0; jj < 32; ++jj) {
      int j = h * 32 + jj;
      float v;
      if (j < 3) v = c3[j];
      else if (j < 63) {
        int t = j - 3, l = t / 6, r = t % 6, d = r % 3;
        float ang = c3[d] * (float)(1 << l);
        v = (r >= 3) ? __cosf(ang) : __sinf(ang);
      } else v = 1.0f;                       // bias row
      sEncP[w][m][j] = (_Float16)v;
    }
  }

  // ---- view encoding (L=4): slots 0..26, slot 27 = 1.0 (bias row), 28..31 = 0 ----
  {
    int m = ln, p = p0 + m;
    float r0 = P.ray_dir[p * 3 + 0], r1 = P.ray_dir[p * 3 + 1], r2 = P.ray_dir[p * 3 + 2];
    float inv = rsqrtf(r0 * r0 + r1 * r1 + r2 * r2);
    float v3[3] = {r0 * inv, r1 * inv, r2 * inv};
#pragma unroll
    for (int jj = 0; jj < 16; ++jj) {
      int j = h * 16 + jj;
      float v;
      if (j < 3) v = v3[j];
      else if (j < 27) {
        int t = j - 3, l = t / 6, r = t % 6, d = r % 3;
        float ang = v3[d] * (float)(1 << l);
        v = (r >= 3) ? __cosf(ang) : __sinf(ang);
      } else v = (j == 27) ? 1.0f : 0.0f;    // bias row + zero pad
      sEncV[w][m][j] = (_Float16)v;
    }
  }
  __syncthreads();

  // ---- net = [enc|1] @ [W_in;b] + condS @ W_c  (biases folded in frag row) ----
  v8f C[8];
#pragma unroll
  for (int t = 0; t < 8; ++t) C[t] = (v8f){};
  {
    v16h aE[2], aC[4];
#pragma unroll
    for (int c = 0; c < 2; ++c) aE[c] = load_A_lds(&sEncP[w][0][0], 64, c * 32, ln, h);
#pragma unroll
    for (int c = 0; c < 4; ++c) aC[c] = load_A_lds(&sCondS[w][0][0], HH, c * 32, ln, h);
#pragma unroll
    for (int t = 0; t < 8; ++t) {
#pragma unroll
      for (int c = 0; c < 2; ++c)
        C[t] = WMMA_F16(aE[c], load_B_ws(P.ws + OFF_WIN, t, c, 2, lane), C[t]);
#pragma unroll
      for (int c = 0; c < 4; ++c)
        C[t] = WMMA_F16(aC[c], load_B_ws(P.ws + OFF_WC, t, c, 4, lane), C[t]);
    }
  }

  // ---- 8 hidden blocks ----
  for (int idx = 0; idx < NBLK; ++idx) {
    const _Float16* fragWb = P.ws + OFF_WBLK + (size_t)idx * 16384;
    if (idx + 1 < NBLK) __builtin_prefetch(fragWb + 16384, 0, 0);  // global_prefetch_b8

    if (idx == 4) {  // skip: += [enc|1]@[W_skip;b] + condS@W_skipc (biases folded)
      v16h aE[2], aC[4];
#pragma unroll
      for (int c = 0; c < 2; ++c) aE[c] = load_A_lds(&sEncP[w][0][0], 64, c * 32, ln, h);
#pragma unroll
      for (int c = 0; c < 4; ++c) aC[c] = load_A_lds(&sCondS[w][0][0], HH, c * 32, ln, h);
#pragma unroll
      for (int t = 0; t < 8; ++t) {
#pragma unroll
        for (int c = 0; c < 2; ++c)
          C[t] = WMMA_F16(aE[c], load_B_ws(P.ws + OFF_WSKIP, t, c, 2, lane), C[t]);
#pragma unroll
        for (int c = 0; c < 4; ++c)
          C[t] = WMMA_F16(aC[c], load_B_ws(P.ws + OFF_WSKIPC, t, c, 4, lane), C[t]);
      }
    }

    // relu(net) -> f16 LDS tile (C-frag layout: m = r + 8h, n = t*16 + ln)
#pragma unroll
    for (int t = 0; t < 8; ++t)
#pragma unroll
      for (int r = 0; r < 8; ++r)
        sAct[w][r + 8 * h][t * 16 + ln] = (_Float16)fmaxf(C[t][r], 0.0f);
    __syncthreads();

    v16h aA[4];
#pragma unroll
    for (int c = 0; c < 4; ++c) aA[c] = load_A_lds(&sAct[w][0][0], HH, c * 32, ln, h);
#pragma unroll
    for (int t = 0; t < 8; ++t) {
      float bias = P.b_blocks[idx * HH + t * 16 + ln];
#pragma unroll
      for (int r = 0; r < 8; ++r) C[t][r] = bias;
    }
#pragma unroll
    for (int t = 0; t < 8; ++t)
#pragma unroll
      for (int c = 0; c < 4; ++c)
        C[t] = WMMA_F16(aA[c], load_B_ws(fragWb, t, c, 4, lane), C[t]);
    __syncthreads();
  }

  // ---- sigma head: 4 WMMAs on relu(net) with 128x16 padded W_out ----
#pragma unroll
  for (int t = 0; t < 8; ++t)
#pragma unroll
    for (int r = 0; r < 8; ++r)
      sAct[w][r + 8 * h][t * 16 + ln] = (_Float16)fmaxf(C[t][r], 0.0f);
  __syncthreads();
  {
    v16h aA[4];
#pragma unroll
    for (int c = 0; c < 4; ++c) aA[c] = load_A_lds(&sAct[w][0][0], HH, c * 32, ln, h);
    v8f D = (v8f){};
#pragma unroll
    for (int c = 0; c < 4; ++c)
      D = WMMA_F16(aA[c], load_B_ws(P.ws + OFF_WOUT, 0, c, 4, lane), D);
    if (ln == 0) {                       // col 0 = sigma; rows m = r + 8h
#pragma unroll
      for (int r = 0; r < 8; ++r) {
        int p = p0 + r + 8 * h;
        float sig = D[r] + P.b_out[0] + P.noise[p];
        P.out[(size_t)p * 4 + 0] = fmaxf(sig, 0.0f);
      }
    }
  }

  // ---- view branch: net += [encV|1]@[W_view;b] + condA@W_viewc (biases folded) ----
  {
    v16h aV = load_A_lds(&sEncV[w][0][0], 32, 0, ln, h);
    v16h aC[4];
#pragma unroll
    for (int c = 0; c < 4; ++c) aC[c] = load_A_lds(&sCondA[w][0][0], HH, c * 32, ln, h);
#pragma unroll
    for (int t = 0; t < 8; ++t) {
      C[t] = WMMA_F16(aV, load_B_ws(P.ws + OFF_WVIEW, t, 0, 1, lane), C[t]);
#pragma unroll
      for (int c = 0; c < 4; ++c)
        C[t] = WMMA_F16(aC[c], load_B_ws(P.ws + OFF_WVIEWC, t, c, 4, lane), C[t]);
    }
  }

  // ---- rgb head: 4 WMMAs on relu(net) with 128x16 padded W_rgb ----
#pragma unroll
  for (int t = 0; t < 8; ++t)
#pragma unroll
    for (int r = 0; r < 8; ++r)
      sAct[w][r + 8 * h][t * 16 + ln] = (_Float16)fmaxf(C[t][r], 0.0f);
  __syncthreads();
  {
    v16h aA[4];
#pragma unroll
    for (int c = 0; c < 4; ++c) aA[c] = load_A_lds(&sAct[w][0][0], HH, c * 32, ln, h);
    v8f D = (v8f){};
#pragma unroll
    for (int c = 0; c < 4; ++c)
      D = WMMA_F16(aA[c], load_B_ws(P.ws + OFF_WRGB, 0, c, 4, lane), D);
    if (ln < 3) {                        // cols 0..2 = r,g,b; rows m = r + 8h
      float brgb = P.b_rgb[ln];
#pragma unroll
      for (int r = 0; r < 8; ++r) {
        int p = p0 + r + 8 * h;
        float v = D[r] + brgb;
        P.out[(size_t)p * 4 + 1 + ln] = 1.0f / (1.0f + __expf(-v));
      }
    }
  }
}

extern "C" void kernel_launch(void* const* d_in, const int* in_sizes, int n_in,
                              void* d_out, int out_size, void* d_ws, size_t ws_size,
                              hipStream_t stream) {
  (void)n_in; (void)ws_size; (void)out_size;
  const float* W_in    = (const float*)d_in[4];
  const float* b_in    = (const float*)d_in[5];
  const float* W_blocks= (const float*)d_in[6];
  const float* W_c     = (const float*)d_in[8];
  const float* b_c     = (const float*)d_in[9];
  const float* W_skip  = (const float*)d_in[10];
  const float* b_skip  = (const float*)d_in[11];
  const float* W_skipc = (const float*)d_in[12];
  const float* b_skipc = (const float*)d_in[13];
  const float* W_out   = (const float*)d_in[14];
  const float* W_viewc = (const float*)d_in[16];
  const float* b_viewc = (const float*)d_in[17];
  const float* W_view  = (const float*)d_in[18];
  const float* b_view  = (const float*)d_in[19];
  const float* W_rgb   = (const float*)d_in[20];
  _Float16* ws = (_Float16*)d_ws;

  // --- pre-pass: swizzle weights (+ folded biases) into f16 B-fragment order ---
  SwzParams S;
  S.ws = ws;
  S.jobs[0]  = {W_in,    b_in,   b_c,     OFF_WIN,     63, 2, 8, HH, HH};
  S.jobs[1]  = {W_c,     nullptr,nullptr, OFF_WC,     128, 4, 8, HH, HH};
  S.jobs[2]  = {W_skip,  b_skip, b_skipc, OFF_WSKIP,   63, 2, 8, HH, HH};
  S.jobs[3]  = {W_skipc, nullptr,nullptr, OFF_WSKIPC, 128, 4, 8, HH, HH};
  S.jobs[4]  = {W_view,  b_view, b_viewc, OFF_WVIEW,   27, 1, 8, HH, HH};
  S.jobs[5]  = {W_viewc, nullptr,nullptr, OFF_WVIEWC, 128, 4, 8, HH, HH};
  for (int i = 0; i < NBLK; ++i)
    S.jobs[6 + i] = {W_blocks + (size_t)i * HH * HH, nullptr, nullptr,
                     OFF_WBLK + i * 16384, 128, 4, 8, HH, HH};
  S.jobs[14] = {W_out,   nullptr,nullptr, OFF_WOUT,   128, 4, 1, 1, 1};
  S.jobs[15] = {W_rgb,   nullptr,nullptr, OFF_WRGB,   128, 4, 1, 3, 3};
  swizzle_weights_kernel<<<dim3(64, 16), 256, 0, stream>>>(S);

  // --- fused MLP ---
  NerfParams P;
  P.coords    = (const float*)d_in[0];
  P.condition = (const float*)d_in[1];
  P.ray_dir   = (const float*)d_in[2];
  P.noise     = (const float*)d_in[3];
  P.b_blocks  = (const float*)d_in[7];
  P.b_out     = (const float*)d_in[15];
  P.b_rgb     = (const float*)d_in[21];
  P.ws        = ws;
  P.out       = (float*)d_out;

  const int N = in_sizes[0] / 3;            // 262144
  dim3 grid(N / (WAVES * PTSW)), block(WAVES * 32);
  nerf_fused_kernel<<<grid, block, 0, stream>>>(P);
}
